// CrossAttnBlock_16398185136214
// MI455X (gfx1250) — compile-verified, matrix operand order checked by reading, not run on hardware
//
#include <hip/hip_runtime.h>
#include <stdint.h>

typedef __bf16 bf16_t;
typedef __attribute__((ext_vector_type(16))) __bf16 v16bf;
typedef __attribute__((ext_vector_type(8)))  float  v8f;
typedef __attribute__((ext_vector_type(4)))  uint32_t v4u;
typedef int v4i_ __attribute__((__vector_size__(16)));   // matches builtin pointee

union FragU { uint32_t u[8]; v16bf v; };

static __device__ __forceinline__ bf16_t f2bf(float f) {
  union { float f; uint32_t u; } cv; cv.f = f;
  uint32_t u = cv.u;
  uint32_t r = u + 0x7FFFu + ((u >> 16) & 1u);   // round-to-nearest-even
  unsigned short h = (unsigned short)(r >> 16);
  return __builtin_bit_cast(bf16_t, h);
}

// per-lane dword index within a 32-K-wide 16-bit fragment row
static __device__ __forceinline__ int frag_kd(int i, int lhi) {
  return (i & 3) + lhi * 4 + ((i >> 2) << 3);
}

// ---- CDNA5 async global->LDS staging (guarded; sync b128 fallback) --------
#if defined(__AMDGCN__) && __has_builtin(__builtin_amdgcn_global_load_async_to_lds_b128) && __has_builtin(__builtin_amdgcn_s_wait_asynccnt)
#define ATHENA_ASYNC 1
#endif

static __device__ __forceinline__ void stage16(const void* g, uint32_t* l) {
#ifdef ATHENA_ASYNC
  __builtin_amdgcn_global_load_async_to_lds_b128(
      (__attribute__((address_space(1))) v4i_*)g,
      (__attribute__((address_space(3))) v4i_*)l, 0, 0);
#else
  *(v4u*)l = *(const v4u*)g;
#endif
}

static __device__ __forceinline__ void stage_wait() {
#ifdef ATHENA_ASYNC
  __builtin_amdgcn_s_wait_asynccnt(0);
#endif
}

// ---------------------------------------------------------------------------
// fp32 [K][N] -> bf16 transposed [N][K]
// ---------------------------------------------------------------------------
__global__ void cvt_transpose_kernel(const float* __restrict__ w,
                                     bf16_t* __restrict__ wt, int K, int Ncol) {
  int i = blockIdx.x * 256 + threadIdx.x;
  int total = K * Ncol;
  if (i < total) {
    int k = i / Ncol, n = i - k * Ncol;
    wt[(size_t)n * K + k] = f2bf(w[i]);
  }
}

// ---------------------------------------------------------------------------
// LayerNorm + Swish over rows of 512, fp32 in -> bf16 out. 1 block per row.
// ---------------------------------------------------------------------------
__global__ __launch_bounds__(256) void ln_swish_kernel(
    const float* __restrict__ in, const float* __restrict__ g,
    const float* __restrict__ bp, bf16_t* __restrict__ out) {
  const int row = blockIdx.x;
  const float* x = in + (size_t)row * 512;
  const int t = threadIdx.x;
  float v0 = x[t], v1 = x[t + 256];
  float s = v0 + v1, s2 = v0 * v0 + v1 * v1;
#pragma unroll
  for (int d = 16; d; d >>= 1) { s += __shfl_down(s, d); s2 += __shfl_down(s2, d); }
  __shared__ float rs[8], rs2[8];
  __shared__ float mu_s, ri_s;
  int wave = t >> 5, lane = t & 31;
  if (lane == 0) { rs[wave] = s; rs2[wave] = s2; }
  __syncthreads();
  if (t == 0) {
    float S = 0.f, S2 = 0.f;
    for (int i = 0; i < 8; ++i) { S += rs[i]; S2 += rs2[i]; }
    float mu = S * (1.f / 512.f);
    float var = S2 * (1.f / 512.f) - mu * mu;
    mu_s = mu; ri_s = rsqrtf(var + 1e-5f);
  }
  __syncthreads();
  float mu = mu_s, ri = ri_s;
  {
    float y = (v0 - mu) * ri * g[t] + bp[t];
    y = y / (1.f + __expf(-y));
    out[(size_t)row * 512 + t] = f2bf(y);
  }
  {
    int c = t + 256;
    float y = (v1 - mu) * ri * g[c] + bp[c];
    y = y / (1.f + __expf(-y));
    out[(size_t)row * 512 + c] = f2bf(y);
  }
}

// ---------------------------------------------------------------------------
// bf16 GEMM: C[R][Ncols] = A[R][K] @ Wt[Ncols][K]^T + bias.
// 128x128 tile per 128-thread block (4 waves, 2x2 grid, 64x64 per wave).
// K in 64-wide slabs, DOUBLE-BUFFERED through LDS via async-to-LDS so the
// ASYNCcnt DMA of slab ks+1 overlaps the 32 WMMAs of slab ks.
// EPI: 0 = bf16 out, 1 = swish -> bf16 out, 2 = +res(fp32) -> fp32 out
// ---------------------------------------------------------------------------
template <int EPI>
__global__ __launch_bounds__(128) void gemm_bf16_kernel(
    const bf16_t* __restrict__ A, const bf16_t* __restrict__ Wt,
    const float* __restrict__ bias, const float* __restrict__ res,
    void* __restrict__ outv, int R, int K, int Ncols) {
  __shared__ uint32_t At[2][128 * 36];  // 128 rows x 64 halfs (32 dwords) +4 pad
  __shared__ uint32_t Bt[2][128 * 36];
  const int tid = threadIdx.x;
  const int wave = tid >> 5, lane = tid & 31;
  const int lrow = lane & 15, lhi = (lane >> 4) & 1;
  const int wr = wave >> 1, wc = wave & 1;   // 2x2 wave grid: 64x64 per wave
  const int rowBase = blockIdx.x * 128, colBase = blockIdx.y * 128;

  v8f zero = {};
  v8f acc[4][4];
#pragma unroll
  for (int i = 0; i < 4; ++i)
#pragma unroll
    for (int j = 0; j < 4; ++j) acc[i][j] = zero;

  // issue one 64-wide K-slab into buffer `buf` (1024 b128 per matrix)
  auto issue = [&](int k0, int buf) {
#pragma unroll
    for (int i = 0; i < 8; ++i) {
      int t128 = i * 128 + tid;        // 0..1023
      int r = t128 >> 3;               // row 0..127
      int c = (t128 & 7) << 2;         // dword col 0,4,..,28
      stage16(A  + (size_t)(rowBase + r) * K + k0 + c * 2, &At[buf][r * 36 + c]);
      stage16(Wt + (size_t)(colBase + r) * K + k0 + c * 2, &Bt[buf][r * 36 + c]);
    }
  };

  const int nk = K >> 6;               // K is a multiple of 64
  issue(0, 0);
  for (int ks = 0; ks < nk; ++ks) {
    stage_wait();                      // slab ks landed in LDS (this wave)
    __syncthreads();                   // all waves: data visible, prev buf free
    if (ks + 1 < nk) issue((ks + 1) << 6, (ks + 1) & 1);   // DMA next slab
    if (ks + 2 < nk) {                 // prefetch two slabs ahead into L2
      __builtin_prefetch(A  + (size_t)(rowBase + tid) * K + ((ks + 2) << 6), 0, 3);
      __builtin_prefetch(Wt + (size_t)(colBase + tid) * K + ((ks + 2) << 6), 0, 3);
    }
    const uint32_t* Ab = At[ks & 1];
    const uint32_t* Bb = Bt[ks & 1];
#pragma unroll
    for (int s = 0; s < 2; ++s) {      // two K=32 sub-steps
      FragU af[4], bfr[4];
#pragma unroll
      for (int tI = 0; tI < 4; ++tI) {
        int r = wr * 64 + tI * 16 + lrow;
#pragma unroll
        for (int i = 0; i < 8; ++i)
          af[tI].u[i] = Ab[r * 36 + s * 16 + frag_kd(i, lhi)];
      }
#pragma unroll
      for (int tI = 0; tI < 4; ++tI) {
        int r = wc * 64 + tI * 16 + lrow;
#pragma unroll
        for (int i = 0; i < 8; ++i)
          bfr[tI].u[i] = Bb[r * 36 + s * 16 + frag_kd(i, lhi)];
      }
#pragma unroll
      for (int i = 0; i < 4; ++i)
#pragma unroll
        for (int j = 0; j < 4; ++j)
          acc[i][j] = __builtin_amdgcn_wmma_f32_16x16x32_bf16(
              false, af[i].v, false, bfr[j].v, (short)0, acc[i][j], false, false);
    }
  }

#pragma unroll
  for (int i = 0; i < 4; ++i) {
    int row0 = rowBase + wr * 64 + i * 16 + lhi * 8;
#pragma unroll
    for (int j = 0; j < 4; ++j) {
      int col = colBase + wc * 64 + j * 16 + lrow;
      float bc = bias[col];
#pragma unroll
      for (int r = 0; r < 8; ++r) {
        float v = acc[i][j][r] + bc;
        size_t idx = (size_t)(row0 + r) * Ncols + col;
        if (EPI == 0) {
          ((bf16_t*)outv)[idx] = f2bf(v);
        } else if (EPI == 1) {
          v = v / (1.f + __expf(-v));
          ((bf16_t*)outv)[idx] = f2bf(v);
        } else {
          ((float*)outv)[idx] = v + res[idx];
        }
      }
    }
  }
}

// ---------------------------------------------------------------------------
// Fused attention: per WG one (b,h) and 16 query rows; S kept in registers.
// qb: [B*N][512] (h*64+d), kvb: [B*M][4608] (k at h*64, v at 512+h*512),
// outp: [B*N][4096] (h*512+c), softmax rows over M=1024.
// ---------------------------------------------------------------------------
__global__ __launch_bounds__(256) void attn_kernel(
    const bf16_t* __restrict__ qb, const bf16_t* __restrict__ kvb,
    bf16_t* __restrict__ outp) {
  constexpr int AN = 1024, AM = 1024;
  const int ntile = blockIdx.x, h = blockIdx.y, b = blockIdx.z;
  const int tid = threadIdx.x, wave = tid >> 5, lane = tid & 31;
  const int lrow = lane & 15, lhi = (lane >> 4) & 1;

  __shared__ uint32_t Pb[16 * 262];   // P bf16 [16][524-half stride], 16.8 KB
  __shared__ uint32_t Vt[512 * 18];   // V^T bf16 [512][36-half stride], 36.9 KB
  __shared__ float wred[8][16];
  __shared__ float gmax[16];
  __shared__ float ginv[16];

  // ---- Q fragments (rows n0..n0+15, d = 0..63 in two K=32 chunks) ----
  FragU qf[2];
  {
    const bf16_t* qrow = qb + (size_t)(b * AN + ntile * 16 + lrow) * 512 + h * 64;
#pragma unroll
    for (int t = 0; t < 2; ++t)
#pragma unroll
      for (int i = 0; i < 8; ++i)
        qf[t].u[i] = *(const uint32_t*)(qrow + t * 32 + frag_kd(i, lhi) * 2);
  }

  // ---- S = Q @ K^T ; wave w owns cols [w*128, w*128+128) -> 8 tiles ----
  v8f zero = {};
  v8f S[8];
#pragma unroll
  for (int j = 0; j < 8; ++j) S[j] = zero;
#pragma unroll
  for (int j = 0; j < 8; ++j) {
    int m0 = wave * 128 + j * 16;
    const bf16_t* krow = kvb + (size_t)(b * AM + m0 + lrow) * 4608 + h * 64;
    FragU kf0, kf1;
#pragma unroll
    for (int i = 0; i < 8; ++i) {
      int kd = frag_kd(i, lhi) * 2;
      kf0.u[i] = *(const uint32_t*)(krow + kd);
      kf1.u[i] = *(const uint32_t*)(krow + 32 + kd);
    }
    S[j] = __builtin_amdgcn_wmma_f32_16x16x32_bf16(false, qf[0].v, false, kf0.v,
                                                   (short)0, S[j], false, false);
    S[j] = __builtin_amdgcn_wmma_f32_16x16x32_bf16(false, qf[1].v, false, kf1.v,
                                                   (short)0, S[j], false, false);
  }

  // ---- row max (shuffle within 16-lane half, then LDS across waves) ----
  float mx[8];
#pragma unroll
  for (int r = 0; r < 8; ++r) {
    float m = S[0][r];
#pragma unroll
    for (int j = 1; j < 8; ++j) m = fmaxf(m, S[j][r]);
#pragma unroll
    for (int d = 1; d < 16; d <<= 1) m = fmaxf(m, __shfl_xor(m, d));
    mx[r] = m;
  }
  if ((lane & 15) == 0) {
#pragma unroll
    for (int r = 0; r < 8; ++r) wred[wave][lhi * 8 + r] = mx[r];
  }
  __syncthreads();
  if (tid < 16) {
    float m = wred[0][tid];
    for (int w = 1; w < 8; ++w) m = fmaxf(m, wred[w][tid]);
    gmax[tid] = m;
  }
  __syncthreads();

  // ---- exp((s - smax)*scale) in place, row sums ----
  const float sl2e = 0.125f * 1.44269504088896f;   // scale * log2(e)
  float sm[8];
#pragma unroll
  for (int r = 0; r < 8; ++r) {
    float gm = gmax[lhi * 8 + r];
    float s = 0.f;
#pragma unroll
    for (int j = 0; j < 8; ++j) {
      float e = exp2f((S[j][r] - gm) * sl2e);
      S[j][r] = e;
      s += e;
    }
#pragma unroll
    for (int d = 1; d < 16; d <<= 1) s += __shfl_xor(s, d);
    sm[r] = s;
  }
  if ((lane & 15) == 0) {
#pragma unroll
    for (int r = 0; r < 8; ++r) wred[wave][lhi * 8 + r] = sm[r];
  }
  __syncthreads();
  if (tid < 16) {
    float s = 0.f;
    for (int w = 0; w < 8; ++w) s += wred[w][tid];
    ginv[tid] = 1.0f / s;
  }
  __syncthreads();

  // ---- O = P @ V  (wave w owns out cols [w*64, w*64+64) -> 4 tiles) ----
  v8f O[4];
#pragma unroll
  for (int j = 0; j < 4; ++j) O[j] = zero;

  bf16_t* Pbh = (bf16_t*)Pb;
  bf16_t* Vth = (bf16_t*)Vt;

  for (int half = 0; half < 2; ++half) {
    // waves owning this half of S write P to LDS (C-layout -> row-major)
    if ((wave >> 2) == half) {
      int cbase = (wave & 3) * 128 + lrow;
#pragma unroll
      for (int j = 0; j < 8; ++j)
#pragma unroll
        for (int r = 0; r < 8; ++r)
          Pbh[(r + lhi * 8) * 524 + cbase + j * 16] = f2bf(S[j][r]);
    }
    __syncthreads();
    for (int mc = 0; mc < 16; ++mc) {
      int m0 = b * AM + half * 512 + mc * 32;
      // stage V[32 rows][512 cols] transposed into Vt[c][m]
#pragma unroll 4
      for (int i = 0; i < 32; ++i) {
        int dv = i * 256 + tid;
        int mr = dv >> 8;          // 0..31
        int c2 = dv & 255;         // dword column
        uint32_t w = *(const uint32_t*)(kvb + (size_t)(m0 + mr) * 4608 +
                                        512 + h * 512 + c2 * 2);
        Vth[(c2 * 2 + 0) * 36 + mr] = __builtin_bit_cast(bf16_t, (unsigned short)(w & 0xFFFFu));
        Vth[(c2 * 2 + 1) * 36 + mr] = __builtin_bit_cast(bf16_t, (unsigned short)(w >> 16));
      }
      __syncthreads();
      FragU pf;
#pragma unroll
      for (int i = 0; i < 8; ++i)
        pf.u[i] = Pb[lrow * 262 + mc * 16 + frag_kd(i, lhi)];
#pragma unroll
      for (int j = 0; j < 4; ++j) {
        FragU vf;
        int c = wave * 64 + j * 16 + lrow;
#pragma unroll
        for (int i = 0; i < 8; ++i) vf.u[i] = Vt[c * 18 + frag_kd(i, lhi)];
        O[j] = __builtin_amdgcn_wmma_f32_16x16x32_bf16(false, pf.v, false, vf.v,
                                                       (short)0, O[j], false, false);
      }
      __syncthreads();
    }
  }

  // ---- normalize by row sum, store bf16 ----
#pragma unroll
  for (int j = 0; j < 4; ++j) {
    int col = wave * 64 + j * 16 + lrow;
#pragma unroll
    for (int r = 0; r < 8; ++r) {
      int row = r + lhi * 8;
      int n = ntile * 16 + row;
      float v = O[j][r] * ginv[row];
      outp[(size_t)(b * AN + n) * 4096 + h * 512 + col] = f2bf(v);
    }
  }
}

// ---------------------------------------------------------------------------
// Host launcher
// ---------------------------------------------------------------------------
extern "C" void kernel_launch(void* const* d_in, const int* in_sizes, int n_in,
                              void* d_out, int out_size, void* d_ws, size_t ws_size,
                              hipStream_t stream) {
  (void)in_sizes; (void)n_in; (void)out_size; (void)ws_size;
  constexpr int B = 8, N = 1024, M = 1024, DIN = 512, H = 8, DEX = 2048;
  constexpr int ROWS = B * N;          // 8192
  constexpr int CROWS = B * M;         // 8192

  const float* x        = (const float*)d_in[0];
  const float* context  = (const float*)d_in[1];
  const float* q_g      = (const float*)d_in[2];
  const float* q_b      = (const float*)d_in[3];
  const float* q_w      = (const float*)d_in[4];
  const float* q_bias   = (const float*)d_in[5];
  const float* kv_g     = (const float*)d_in[6];
  const float* kv_b     = (const float*)d_in[7];
  const float* kv_w     = (const float*)d_in[8];
  const float* kv_bias  = (const float*)d_in[9];
  const float* merge_w  = (const float*)d_in[10];
  const float* merge_b  = (const float*)d_in[11];
  const float* ff_g     = (const float*)d_in[12];
  const float* ff_b     = (const float*)d_in[13];
  const float* ff_w1    = (const float*)d_in[14];
  const float* ff_b1    = (const float*)d_in[15];
  const float* ff_w2    = (const float*)d_in[16];
  const float* ff_b2    = (const float*)d_in[17];
  float* out = (float*)d_out;

  char* ws = (char*)d_ws;
  size_t off = 0;
  auto alloc = [&](size_t bytes) -> char* {
    char* p = ws + off;
    off = (off + bytes + 255) & ~(size_t)255;
    return p;
  };
  bf16_t* xa      = (bf16_t*)alloc((size_t)ROWS * DIN * 2);
  bf16_t* ca      = (bf16_t*)alloc((size_t)CROWS * DIN * 2);
  bf16_t* qb      = (bf16_t*)alloc((size_t)ROWS * 512 * 2);
  bf16_t* kvb     = (bf16_t*)alloc((size_t)CROWS * 4608 * 2);
  bf16_t* ao      = (bf16_t*)alloc((size_t)ROWS * 4096 * 2);
  float*  x2      = (float*)alloc((size_t)ROWS * DIN * 4);
  bf16_t* ffa     = (bf16_t*)alloc((size_t)ROWS * DIN * 2);
  bf16_t* hb      = (bf16_t*)alloc((size_t)ROWS * DEX * 2);
  bf16_t* q_wt    = (bf16_t*)alloc((size_t)512 * 512 * 2);
  bf16_t* kv_wt   = (bf16_t*)alloc((size_t)4608 * 512 * 2);
  bf16_t* merge_wt= (bf16_t*)alloc((size_t)512 * 4096 * 2);
  bf16_t* ff_w1t  = (bf16_t*)alloc((size_t)2048 * 512 * 2);
  bf16_t* ff_w2t  = (bf16_t*)alloc((size_t)512 * 2048 * 2);

  auto tlaunch = [&](const float* w, bf16_t* wt, int K, int Ncol) {
    int total = K * Ncol;
    cvt_transpose_kernel<<<(total + 255) / 256, 256, 0, stream>>>(w, wt, K, Ncol);
  };
  tlaunch(q_w, q_wt, 512, 512);
  tlaunch(kv_w, kv_wt, 512, 4608);
  tlaunch(merge_w, merge_wt, 4096, 512);
  tlaunch(ff_w1, ff_w1t, 512, 2048);
  tlaunch(ff_w2, ff_w2t, 2048, 512);

  ln_swish_kernel<<<ROWS, 256, 0, stream>>>(x, q_g, q_b, xa);
  ln_swish_kernel<<<CROWS, 256, 0, stream>>>(context, kv_g, kv_b, ca);

  gemm_bf16_kernel<0><<<dim3(ROWS / 128, 512 / 128), 128, 0, stream>>>(
      xa, q_wt, q_bias, nullptr, qb, ROWS, 512, 512);
  gemm_bf16_kernel<0><<<dim3(CROWS / 128, 4608 / 128), 128, 0, stream>>>(
      ca, kv_wt, kv_bias, nullptr, kvb, CROWS, 512, 4608);

  attn_kernel<<<dim3(N / 16, H, B), 256, 0, stream>>>(qb, kvb, ao);

  gemm_bf16_kernel<2><<<dim3(ROWS / 128, 512 / 128), 128, 0, stream>>>(
      ao, merge_wt, merge_b, x, x2, ROWS, 4096, 512);

  ln_swish_kernel<<<ROWS, 256, 0, stream>>>(x2, ff_g, ff_b, ffa);

  gemm_bf16_kernel<1><<<dim3(ROWS / 128, DEX / 128), 128, 0, stream>>>(
      ffa, ff_w1t, ff_b1, nullptr, hb, ROWS, 512, DEX);

  gemm_bf16_kernel<2><<<dim3(ROWS / 128, 512 / 128), 128, 0, stream>>>(
      hb, ff_w2t, ff_b2, x2, out, ROWS, DEX, 512);
}
